// EquivariantCrossAttention_38663295598614
// MI455X (gfx1250) — matrix operand order repaired and checked by reading, not compiled, not run on hardware
//
#include <hip/hip_runtime.h>

#define B_ 2
#define N_ 2048
#define L_ 1024
#define K_ 16
#define H_ 4
#define DH_ 128
#define HD_ 512
#define M_ (B_ * N_ * K_)   /* 65536 gathered rows   */
#define R_ (B_ * N_)        /* 4096 query rows       */

typedef __bf16 bf16;
typedef __attribute__((ext_vector_type(16))) __bf16 v16bf;
typedef __attribute__((ext_vector_type(8)))  float  v8f;

#define TWO_PI_F 12.566370614359172f   /* 2*pi*freq, freq = 2.0 for both q and v */
#define ATT_SCALE 0.08838834764831845f /* 1/sqrt(128) */

__device__ __forceinline__ float gelu_f(float x) {
  const float c0 = 0.7978845608028654f; /* sqrt(2/pi), tanh-approx gelu (jax default) */
  return 0.5f * x * (1.f + tanhf(c0 * (x + 0.044715f * x * x * x)));
}

/* ------------------------------------------------------------------ */
/* Top-K nearest latents per (b,n); one wave32 per row, LDS merge.    */
/* ------------------------------------------------------------------ */
__global__ void topk_kernel(const float* __restrict__ x, const float* __restrict__ p,
                            const float* __restrict__ sigma,
                            int* __restrict__ nearest, float* __restrict__ inv,
                            float* __restrict__ gw) {
  __shared__ float sd[4 * 512];
  __shared__ int   si[4 * 512];
  const int wave = threadIdx.x >> 5, lane = threadIdx.x & 31;
  const int r = blockIdx.x * 4 + wave;           /* (b,n) flat row */
  const int b = r / N_;
  const float x0 = x[(size_t)r * 2 + 0];
  const float x1 = x[(size_t)r * 2 + 1];

  float dl[16]; int il[16];
#pragma unroll
  for (int i = 0; i < 16; ++i) { dl[i] = 3.4e38f; il[i] = 0x7fffffff; }
  for (int l = lane; l < L_; l += 32) {
    const float dx = x0 - p[((size_t)b * L_ + l) * 2 + 0];
    const float dy = x1 - p[((size_t)b * L_ + l) * 2 + 1];
    const float d2 = dx * dx + dy * dy;          /* sqrt monotone: rank by d^2 */
    if (d2 < dl[15] || (d2 == dl[15] && l < il[15])) {
      int j = 15;
      while (j > 0 && (d2 < dl[j - 1] || (d2 == dl[j - 1] && l < il[j - 1]))) {
        dl[j] = dl[j - 1]; il[j] = il[j - 1]; --j;
      }
      dl[j] = d2; il[j] = l;
    }
  }
  float* swd = sd + wave * 512; int* swi = si + wave * 512;
#pragma unroll
  for (int i = 0; i < 16; ++i) { swd[lane * 16 + i] = dl[i]; swi[lane * 16 + i] = il[i]; }
  __syncthreads();
  if (lane == 0) {                                /* serial merge of 512 candidates */
    float fd[16]; int fi[16];
#pragma unroll
    for (int i = 0; i < 16; ++i) { fd[i] = 3.4e38f; fi[i] = 0x7fffffff; }
    for (int t = 0; t < 512; ++t) {
      const float d2 = swd[t]; const int l = swi[t];
      if (d2 < fd[15] || (d2 == fd[15] && l < fi[15])) {
        int j = 15;
        while (j > 0 && (d2 < fd[j - 1] || (d2 == fd[j - 1] && l < fi[j - 1]))) {
          fd[j] = fd[j - 1]; fi[j] = fi[j - 1]; --j;
        }
        fd[j] = d2; fi[j] = l;
      }
    }
    for (int k = 0; k < K_; ++k) {
      const int li = fi[k];
      const size_t m = (size_t)r * K_ + k;
      nearest[m] = li;
      inv[m * 2 + 0] = x0 - p[((size_t)b * L_ + li) * 2 + 0];
      inv[m * 2 + 1] = x1 - p[((size_t)b * L_ + li) * 2 + 1];
      const float sg = sigma[(size_t)b * L_ + li];
      gw[m] = -0.5f * fd[k] / (sg * sg);
    }
  }
}

/* ------------------------------------------------------------------ */
/* RFF features (sin|cos) for q and v paths + gather of c rows (bf16) */
/* ------------------------------------------------------------------ */
__global__ void embed_kernel(const float* __restrict__ c, const float* __restrict__ rffq,
                             const float* __restrict__ rffv, const int* __restrict__ nearest,
                             const float* __restrict__ inv, bf16* __restrict__ cg,
                             bf16* __restrict__ fq, bf16* __restrict__ fv) {
  const size_t t = (size_t)blockIdx.x * blockDim.x + threadIdx.x; /* m*128 + d */
  const size_t m = t >> 7; const int d = (int)(t & 127);
  const int b = (int)(m / ((size_t)N_ * K_));
  const int li = nearest[m];
  cg[t] = (bf16)c[((size_t)b * L_ + li) * DH_ + d];
  const float i0 = inv[m * 2], i1 = inv[m * 2 + 1];
  const int j = d & 63;
  const float pq = TWO_PI_F * (i0 * rffq[j] + i1 * rffq[64 + j]);
  const float pv = TWO_PI_F * (i0 * rffv[j] + i1 * rffv[64 + j]);
  fq[t] = (bf16)((d >= 64) ? cosf(pq) : sinf(pq));
  fv[t] = (bf16)((d >= 64) ? cosf(pv) : sinf(pv));
}

/* f32 weight (Kd x Nd) -> bf16 transposed (Nd x Kd); read coalesced  */
__global__ void wconv_kernel(const float* __restrict__ W, bf16* __restrict__ Wt,
                             int Kd, int Nd) {
  const int t = blockIdx.x * 256 + threadIdx.x;
  if (t >= Kd * Nd) return;
  const int k = t / Nd, nc = t % Nd;
  Wt[(size_t)nc * Kd + k] = (bf16)W[t];
}

/* ------------------------------------------------------------------ */
/* bf16 WMMA GEMM: C = act(A[M x Kd] @ W[Kd x Nd] + bias).            */
/* Block tile 64x64; 4 waves, each wave = 64 rows x 16 cols strip:    */
/* one B fragment (2x global b128 from transposed L2-resident bf16    */
/* weights) reused across 4 A fragments (2x ds b128 each) -> 4 WMMA.  */
/* A tile staged via GLOBAL_LOAD_ASYNC_TO_LDS_B128 (ASYNCcnt DMA,     */
/* no VGPR round-trip). Fragment layouts per ISA 7.12.2.              */
/* ------------------------------------------------------------------ */
template <int ACT, bool OUT_BF16>
__global__ void gemm_bf16_kernel(const bf16* __restrict__ A, const bf16* __restrict__ Wt,
                                 const float* __restrict__ bias, void* __restrict__ Cout,
                                 int Mrows, int Kd, int Nd) {
  __shared__ __align__(16) bf16 As[64 * 32];
  const int tid = threadIdx.x;
  const int wave = tid >> 5, lane = tid & 31;
  const int halfId = lane >> 4, l16 = lane & 15;
  const int mBase = blockIdx.x * 64;
  const int myCol = blockIdx.y * 64 + wave * 16 + l16;

  /* per-thread staging: thread owns rows rr0 and rr0+32, one 16B chunk each */
  const int rr0 = tid >> 2, cc0 = (tid & 3) * 8;
  const bf16* ag0 = &A[(size_t)(mBase + rr0) * Kd + cc0];
  const bf16* ag1 = &A[(size_t)(mBase + rr0 + 32) * Kd + cc0];
  const unsigned l0 = (unsigned)(uintptr_t)&As[rr0 * 32 + cc0];
  const unsigned l1 = (unsigned)(uintptr_t)&As[(rr0 + 32) * 32 + cc0];

  union BF16x16 { v16bf v; uint4 q[2]; };
  v8f acc[4] = {v8f{}, v8f{}, v8f{}, v8f{}};

  for (int k0 = 0; k0 < Kd; k0 += 32) {
    /* async DMA 64x32 A tile straight into LDS (no VGPR data path) */
    {
      const unsigned long long g0 = (unsigned long long)(const void*)(ag0 + k0);
      const unsigned long long g1 = (unsigned long long)(const void*)(ag1 + k0);
      asm volatile("global_load_async_to_lds_b128 %0, %1, off"
                   :: "v"(l0), "v"(g0) : "memory");
      asm volatile("global_load_async_to_lds_b128 %0, %1, off"
                   :: "v"(l1), "v"(g1) : "memory");
      asm volatile("s_wait_asynccnt 0x0" ::: "memory");
    }
    __syncthreads();
    BF16x16 bu;                                   /* B frag: lane col fixed, K split 0..7 / 16..23 (+8 hi half) */
    {
      const bf16* wp = &Wt[(size_t)myCol * Kd + k0 + halfId * 8];
      bu.q[0] = *(const uint4*)wp;
      bu.q[1] = *(const uint4*)(wp + 16);
    }
#pragma unroll
    for (int t = 0; t < 4; ++t) {
      const int row = t * 16 + l16;
      BF16x16 au;                                 /* A frag: K split 0..7 / 16..23 (+8 hi half) */
      au.q[0] = *(const uint4*)&As[row * 32 + halfId * 8];
      au.q[1] = *(const uint4*)&As[row * 32 + 16 + halfId * 8];
      acc[t] = __builtin_amdgcn_wmma_f32_16x16x32_bf16(false, au.v, false, bu.v, (short)0,
                                                       acc[t], false, false);
    }
    __syncthreads();
  }
  const float bv = bias[myCol];
#pragma unroll
  for (int t = 0; t < 4; ++t) {
#pragma unroll
    for (int i = 0; i < 8; ++i) {                 /* C/D: VGPR i -> row i / i+8 per half */
      const int row = mBase + t * 16 + i + halfId * 8;
      float v = acc[t][i] + bv;
      if (ACT == 1) v = gelu_f(v);
      if (OUT_BF16) ((bf16*)Cout)[(size_t)row * Nd + myCol] = (bf16)v;
      else          ((float*)Cout)[(size_t)row * Nd + myCol] = v;
    }
  }
}

/* ------------------------------------------------------------------ */
/* Wave-per-row LayerNorm over 128 dims (bf16 in -> bf16 out).        */
/* ------------------------------------------------------------------ */
__global__ void layernorm_kernel(const bf16* __restrict__ in, const float* __restrict__ s,
                                 const float* __restrict__ b, bf16* __restrict__ out,
                                 int rows) {
  const int wave = threadIdx.x >> 5, lane = threadIdx.x & 31;
  const long row = (long)blockIdx.x * 8 + wave;
  if (row >= rows) return;
  const bf16* rp = in + (size_t)row * 128;
  float v[4]; float sum = 0.f;
#pragma unroll
  for (int i = 0; i < 4; ++i) { v[i] = (float)rp[lane * 4 + i]; sum += v[i]; }
#pragma unroll
  for (int m = 16; m > 0; m >>= 1) sum += __shfl_xor(sum, m, 32);
  const float mean = sum * (1.f / 128.f);
  float var = 0.f;
#pragma unroll
  for (int i = 0; i < 4; ++i) { const float d = v[i] - mean; var += d * d; }
#pragma unroll
  for (int m = 16; m > 0; m >>= 1) var += __shfl_xor(var, m, 32);
  const float rstd = rsqrtf(var * (1.f / 128.f) + 1e-6f);
#pragma unroll
  for (int i = 0; i < 4; ++i) {
    const int dch = lane * 4 + i;
    out[(size_t)row * 128 + dch] = (bf16)((v[i] - mean) * rstd * s[dch] + b[dch]);
  }
}

__global__ void cast_bf16_kernel(const float* __restrict__ in, bf16* __restrict__ out,
                                 size_t n) {
  const size_t t = (size_t)blockIdx.x * blockDim.x + threadIdx.x;
  if (t < n) out[t] = (bf16)in[t];
}

/* inv_emb_v * (1+g) + bt, FiLM from gb per (b,n) row */
__global__ void mod_ev_kernel(const bf16* __restrict__ ev, const float* __restrict__ gb,
                              bf16* __restrict__ out) {
  const size_t t = (size_t)blockIdx.x * blockDim.x + threadIdx.x; /* m*128+d */
  const size_t m = t >> 7; const int d = (int)(t & 127);
  const size_t rr = m >> 4;                                       /* / K */
  const float g = gb[rr * 256 + d], bt = gb[rr * 256 + 128 + d];
  out[t] = (bf16)((float)ev[t] * (1.f + g) + bt);
}

/* v0 * (1+vg) + vb, vg/vb from vgb per gathered row */
__global__ void mod_v_kernel(const bf16* __restrict__ v0, const bf16* __restrict__ vgb,
                             bf16* __restrict__ out) {
  const size_t t = (size_t)blockIdx.x * blockDim.x + threadIdx.x; /* m*512+d */
  const size_t m = t >> 9; const int d = (int)(t & 511);
  const float vg = (float)vgb[m * 1024 + d], vb = (float)vgb[m * 1024 + 512 + d];
  out[t] = (bf16)((float)v0[t] * (1.f + vg) + vb);
}

/* per-(b,n): qk dots per (k,h) + window, softmax over K */
__global__ void attn_kernel(const bf16* __restrict__ q, const bf16* __restrict__ k,
                            const float* __restrict__ gw, float* __restrict__ att) {
  __shared__ float s[64];
  const int r = blockIdx.x, t = threadIdx.x;       /* 64 threads = (k,h) pairs */
  const int kk = t >> 2, h = t & 3;
  const size_t m = (size_t)r * K_ + kk;
  const bf16* qr = q + m * HD_ + h * DH_;
  const bf16* kr = k + m * HD_ + h * DH_;
  float dot = 0.f;
  for (int d = 0; d < DH_; ++d) dot += (float)qr[d] * (float)kr[d];
  s[t] = dot * ATT_SCALE + gw[m];
  __syncthreads();
  float mx = -3.4e38f;
#pragma unroll
  for (int j = 0; j < K_; ++j) mx = fmaxf(mx, s[j * 4 + h]);
  float sum = 0.f;
#pragma unroll
  for (int j = 0; j < K_; ++j) sum += expf(s[j * 4 + h] - mx);
  att[(size_t)r * 64 + t] = expf(s[t] - mx) / sum;
}

/* y[r, hd] = sum_k att[r,k,h] * v2[r,k,hd] */
__global__ void ysum_kernel(const float* __restrict__ att, const bf16* __restrict__ v2,
                            bf16* __restrict__ y) {
  const size_t t = (size_t)blockIdx.x * blockDim.x + threadIdx.x; /* r*512+hd */
  const size_t r = t >> 9; const int hd = (int)(t & 511); const int h = hd >> 7;
  float acc = 0.f;
#pragma unroll
  for (int kk = 0; kk < K_; ++kk)
    acc += att[r * 64 + kk * 4 + h] * (float)v2[((size_t)r * K_ + kk) * HD_ + hd];
  y[t] = (bf16)acc;
}

/* ------------------------------------------------------------------ */
static inline void launch_gemm(const bf16* A, const bf16* Wt, const float* bias, void* C,
                               int Mrows, int Kd, int Nd, int act, bool outbf,
                               hipStream_t s) {
  dim3 grid(Mrows / 64, Nd / 64), block(128);
  if (act) {
    if (outbf) gemm_bf16_kernel<1, true><<<grid, block, 0, s>>>(A, Wt, bias, C, Mrows, Kd, Nd);
    else       gemm_bf16_kernel<1, false><<<grid, block, 0, s>>>(A, Wt, bias, C, Mrows, Kd, Nd);
  } else {
    if (outbf) gemm_bf16_kernel<0, true><<<grid, block, 0, s>>>(A, Wt, bias, C, Mrows, Kd, Nd);
    else       gemm_bf16_kernel<0, false><<<grid, block, 0, s>>>(A, Wt, bias, C, Mrows, Kd, Nd);
  }
}

extern "C" void kernel_launch(void* const* d_in, const int* in_sizes, int n_in,
                              void* d_out, int out_size, void* d_ws, size_t ws_size,
                              hipStream_t stream) {
  (void)in_sizes; (void)n_in; (void)out_size; (void)ws_size;
  const float* x     = (const float*)d_in[0];
  const float* p     = (const float*)d_in[1];
  const float* c     = (const float*)d_in[2];
  const float* wsig  = (const float*)d_in[3];
  const float* x_h   = (const float*)d_in[4];
  const float* rffq  = (const float*)d_in[5];
  const float* rffv  = (const float*)d_in[6];
  const float *eq_w1 = (const float*)d_in[7],  *eq_b1 = (const float*)d_in[8];
  const float *eq_w2 = (const float*)d_in[9],  *eq_b2 = (const float*)d_in[10];
  const float *ev_w1 = (const float*)d_in[11], *ev_b1 = (const float*)d_in[12];
  const float *ev_w2 = (const float*)d_in[13], *ev_b2 = (const float*)d_in[14];
  const float *wq = (const float*)d_in[15], *bq = (const float*)d_in[16];
  const float *wk = (const float*)d_in[17], *bk = (const float*)d_in[18];
  const float *wv = (const float*)d_in[19], *bv = (const float*)d_in[20];
  const float *cw1 = (const float*)d_in[21], *cb1 = (const float*)d_in[22];
  const float *cls = (const float*)d_in[23], *clb = (const float*)d_in[24];
  const float *cw2 = (const float*)d_in[25], *cb2 = (const float*)d_in[26];
  const float *ivw1 = (const float*)d_in[27], *ivb1 = (const float*)d_in[28];
  const float *ivls = (const float*)d_in[29], *ivlb = (const float*)d_in[30];
  const float *ivw2 = (const float*)d_in[31], *ivb2 = (const float*)d_in[32];
  const float *mw1 = (const float*)d_in[33], *mb1 = (const float*)d_in[34];
  const float *mls = (const float*)d_in[35], *mlb = (const float*)d_in[36];
  const float *mw2 = (const float*)d_in[37], *mb2 = (const float*)d_in[38];
  const float *wo = (const float*)d_in[39], *bo = (const float*)d_in[40];

  char* base = (char*)d_ws;
  size_t off = 0;
  auto alloc = [&](size_t bytes) -> void* {
    void* q = base + off;
    off = (off + bytes + 255) & ~(size_t)255;
    return q;
  };
  auto prep = [&](const float* W, int Kd, int Nd) -> bf16* {
    bf16* outW = (bf16*)alloc((size_t)Kd * Nd * 2);
    const int n = Kd * Nd;
    wconv_kernel<<<(n + 255) / 256, 256, 0, stream>>>(W, outW, Kd, Nd);
    return outW;
  };

  int*   nearest = (int*)alloc((size_t)M_ * 4);
  float* invb    = (float*)alloc((size_t)M_ * 2 * 4);
  float* gwb     = (float*)alloc((size_t)M_ * 4);
  bf16*  S1 = (bf16*)alloc((size_t)M_ * 128 * 2);   /* featsq */
  bf16*  S2 = (bf16*)alloc((size_t)M_ * 128 * 2);   /* featsv */
  bf16*  S3 = (bf16*)alloc((size_t)M_ * 128 * 2);   /* cg */
  bf16*  S4 = (bf16*)alloc((size_t)M_ * 128 * 2);   /* gelu hidden / iv1 */
  bf16*  S5 = (bf16*)alloc((size_t)M_ * 128 * 2);   /* eq emb / ev / iv1n */
  bf16*  S6 = (bf16*)alloc((size_t)M_ * 128 * 2);   /* modulated inv_emb_v */
  bf16*  Qb  = (bf16*)alloc((size_t)M_ * HD_ * 2);
  bf16*  Kb  = (bf16*)alloc((size_t)M_ * HD_ * 2);
  bf16*  V0  = (bf16*)alloc((size_t)M_ * HD_ * 2);
  bf16*  VGB = (bf16*)alloc((size_t)M_ * 1024 * 2);
  bf16*  VM  = (bf16*)alloc((size_t)M_ * HD_ * 2);  /* vmod, later m1n */
  bf16*  M1  = (bf16*)alloc((size_t)M_ * HD_ * 2);  /* m1, later v2 */
  bf16*  xh_bf = (bf16*)alloc((size_t)R_ * 128 * 2);
  bf16*  C1  = (bf16*)alloc((size_t)R_ * 128 * 2);
  bf16*  C1n = (bf16*)alloc((size_t)R_ * 128 * 2);
  float* GB  = (float*)alloc((size_t)R_ * 256 * 4);
  float* ATT = (float*)alloc((size_t)R_ * 64 * 4);
  bf16*  Yb  = (bf16*)alloc((size_t)R_ * HD_ * 2);

  /* 0) weights -> bf16 transposed (L2-resident, ~1.1 MB total) */
  bf16* Teq1 = prep(eq_w1, 128, 128);
  bf16* Teq2 = prep(eq_w2, 128, 128);
  bf16* Tev1 = prep(ev_w1, 128, 128);
  bf16* Tev2 = prep(ev_w2, 128, 128);
  bf16* Twq  = prep(wq, 128, HD_);
  bf16* Twk  = prep(wk, 128, HD_);
  bf16* Twv  = prep(wv, 128, HD_);
  bf16* Tcw1 = prep(cw1, 128, 128);
  bf16* Tcw2 = prep(cw2, 128, 256);
  bf16* Tiv1 = prep(ivw1, 128, 128);
  bf16* Tiv2 = prep(ivw2, 128, 1024);
  bf16* Tmw1 = prep(mw1, 128, 128);
  bf16* Tmw2 = prep(mw2, 128, 128);
  bf16* Two  = prep(wo, HD_, 128);

  /* 1) kNN selection + invariants + gaussian window */
  topk_kernel<<<R_ / 4, 128, 0, stream>>>(x, p, wsig, nearest, invb, gwb);
  /* 2) RFF features + c gather */
  embed_kernel<<<(int)((size_t)M_ * 128 / 256), 256, 0, stream>>>(c, rffq, rffv, nearest,
                                                                  invb, S3, S1, S2);
  /* 3) q embedding MLP + projections (all WMMA) */
  launch_gemm(S1, Teq1, eq_b1, S4, M_, 128, 128, 1, true, stream);
  launch_gemm(S4, Teq2, eq_b2, S5, M_, 128, 128, 0, true, stream);
  launch_gemm(S5, Twq, bq, Qb, M_, 128, HD_, 0, true, stream);
  launch_gemm(S3, Twk, bk, Kb, M_, 128, HD_, 0, true, stream);
  launch_gemm(S3, Twv, bv, V0, M_, 128, HD_, 0, true, stream);
  /* 4) v-invariant embedding MLP */
  launch_gemm(S2, Tev1, ev_b1, S4, M_, 128, 128, 1, true, stream);
  launch_gemm(S4, Tev2, ev_b2, S5, M_, 128, 128, 0, true, stream);
  /* 5) conditioning FFN on x_h -> gb */
  cast_bf16_kernel<<<(int)((size_t)R_ * 128 / 256), 256, 0, stream>>>(x_h, xh_bf,
                                                                      (size_t)R_ * 128);
  launch_gemm(xh_bf, Tcw1, cb1, C1, R_, 128, 128, 1, true, stream);
  layernorm_kernel<<<R_ / 8, 256, 0, stream>>>(C1, cls, clb, C1n, R_);
  launch_gemm(C1n, Tcw2, cb2, GB, R_, 128, 256, 0, false, stream);
  /* 6) FiLM then iv FFN -> vgb */
  mod_ev_kernel<<<(int)((size_t)M_ * 128 / 256), 256, 0, stream>>>(S5, GB, S6);
  launch_gemm(S6, Tiv1, ivb1, S4, M_, 128, 128, 1, true, stream);
  layernorm_kernel<<<M_ / 8, 256, 0, stream>>>(S4, ivls, ivlb, S5, M_);
  launch_gemm(S5, Tiv2, ivb2, VGB, M_, 128, 1024, 0, true, stream);
  /* 7) modulate v, per-head FFN (rows = M*H) */
  mod_v_kernel<<<(int)((size_t)M_ * HD_ / 256), 256, 0, stream>>>(V0, VGB, VM);
  launch_gemm(VM, Tmw1, mb1, M1, M_ * 4, 128, 128, 1, true, stream);
  layernorm_kernel<<<(M_ * 4) / 8, 256, 0, stream>>>(M1, mls, mlb, VM, M_ * 4);
  launch_gemm(VM, Tmw2, mb2, M1, M_ * 4, 128, 128, 0, true, stream); /* M1 = v2 */
  /* 8) attention over K + weighted sum + output projection */
  attn_kernel<<<R_, 64, 0, stream>>>(Qb, Kb, gwb, ATT);
  ysum_kernel<<<(int)((size_t)R_ * HD_ / 256), 256, 0, stream>>>(ATT, M1, Yb);
  launch_gemm(Yb, Two, bo, d_out, R_, HD_, 128, 0, false, stream);
}
